// GCNNet_38637525795005
// MI455X (gfx1250) — matrix operand order repaired and checked by reading, not compiled, not run on hardware
//
#include <hip/hip_runtime.h>

typedef __attribute__((ext_vector_type(16))) _Float16 v16h;
typedef __attribute__((ext_vector_type(8)))  _Float16 v8h;
typedef __attribute__((ext_vector_type(8)))  float    v8f;

#define D 128                 // feature dim (D_IN == D_HID)
#define N_GRAPHS 1024
#define D_OUT 2

// ---------------------------------------------------------------------------
// Pack a 128x128 row-major f32 weight matrix into WMMA B-fragment order (f16).
// B fragment (16-bit, 32x16 KxN), contiguous k order used here (same as A): lane l holds column (l&15);
// elements e=0..15 hold K = kbase + (l>>4)*16 + e.
// frag layout: [ntile(8)][kk(4)][lane(32)][e(16)] halfs.
// ---------------------------------------------------------------------------
__global__ void gcn_pack_w(const float* __restrict__ W, _Float16* __restrict__ frag) {
    int t = blockIdx.x * blockDim.x + threadIdx.x;     // 0..1023
    if (t >= 8 * 4 * 32) return;
    int lane  = t & 31;
    int kk    = (t >> 5) & 3;
    int ntile = t >> 7;
    int col = ntile * 16 + (lane & 15);
    int kb  = kk * 32 + (lane >> 4) * 16;
    _Float16* out = frag + (size_t)t * 16;
#pragma unroll
    for (int e = 0; e < 16; ++e)
        out[e] = (_Float16)W[(kb + e) * D + col];
}

// ---------------------------------------------------------------------------
// Degree / normalization
// ---------------------------------------------------------------------------
__global__ void gcn_deg_init(float* __restrict__ deg, int N) {
    int i = blockIdx.x * blockDim.x + threadIdx.x;
    if (i < N) deg[i] = 1.0f;                          // self-loop
}
__global__ void gcn_deg_edges(const int* __restrict__ dst, float* __restrict__ deg, int E) {
    int e = blockIdx.x * blockDim.x + threadIdx.x;
    if (e < E) atomicAdd(&deg[dst[e]], 1.0f);
}
__global__ void gcn_dinv(float* __restrict__ deg, int N) {
    int i = blockIdx.x * blockDim.x + threadIdx.x;
    if (i < N) deg[i] = rsqrtf(deg[i]);                // deg >= 1 always
}

// ---------------------------------------------------------------------------
// WMMA GEMM: C[M x 128] = fp16(A[M x 128]) * fp16(W[128 x 128]), f32 accum.
// One block = 8 waves = one 16-row M tile x all 8 N tiles. M % 16 == 0.
// ---------------------------------------------------------------------------
__global__ __launch_bounds__(256) void gcn_gemm_wmma(
        const float* __restrict__ A, const _Float16* __restrict__ Wfrag,
        float* __restrict__ C) {
    __shared__ _Float16 As[16][136];                   // +8 halfs pad: conflict-free

    int tid = threadIdx.x;
    int m0  = blockIdx.x * 16;

    // Stage A tile (16x128 f32 -> f16 in LDS). thread t: row t>>4, 8 cols.
    {
        int row = tid >> 4;
        int c0  = (tid & 15) * 8;
        const float4* ap = (const float4*)(A + (size_t)(m0 + row) * D + c0);
        float4 a0 = ap[0];
        float4 a1 = ap[1];
        _Float16* sp = &As[row][c0];
        sp[0] = (_Float16)a0.x; sp[1] = (_Float16)a0.y;
        sp[2] = (_Float16)a0.z; sp[3] = (_Float16)a0.w;
        sp[4] = (_Float16)a1.x; sp[5] = (_Float16)a1.y;
        sp[6] = (_Float16)a1.z; sp[7] = (_Float16)a1.w;
    }
    __syncthreads();

    int wave = tid >> 5;          // N tile 0..7
    int lane = tid & 31;
    int arow = lane & 15;
    int half = lane >> 4;

    v8f acc = {};
#pragma unroll
    for (int kk = 0; kk < 4; ++kk) {
        int kb = kk * 32;
        // A fragment (16-bit 16x32): elems 0..7 -> K = kb + half*16 + e,
        //                            elems 8..15 -> K = kb + half*16 + 8 + e.
        v8h lo = *(const v8h*)&As[arow][kb + half * 16];
        v8h hi = *(const v8h*)&As[arow][kb + half * 16 + 8];
        v16h a;
#pragma unroll
        for (int e = 0; e < 8; ++e) { a[e] = lo[e]; a[e + 8] = hi[e]; }
        // B fragment: pre-packed, one 32B load.
        v16h b = *(const v16h*)(Wfrag + ((size_t)(wave * 4 + kk) * 32 + lane) * 16);
        acc = __builtin_amdgcn_wmma_f32_16x16x32_f16(
                  false, a, false, b, (short)0, acc, false, false);
    }

    // D layout: VGPR r -> row = 8*half + r, col = lane&15 (within N tile).
    int col   = wave * 16 + (lane & 15);
    int rbase = half * 8;
#pragma unroll
    for (int r = 0; r < 8; ++r)
        C[(size_t)(m0 + rbase + r) * D + col] = acc[r];
}

// ---------------------------------------------------------------------------
// agg[i] = lin[i] * dinv[i]^2   (self-loop message, also initializes agg)
// ---------------------------------------------------------------------------
__global__ void gcn_selfloop_init(const float* __restrict__ lin,
                                  const float* __restrict__ dinv,
                                  float* __restrict__ agg, int NE) {
    int idx = blockIdx.x * blockDim.x + threadIdx.x;
    if (idx >= NE) return;
    float di = dinv[idx >> 7];
    agg[idx] = lin[idx] * di * di;
}

// ---------------------------------------------------------------------------
// One wave per edge: gather 128-float row of src (float4/lane), scale by
// dinv[src]*dinv[dst], atomic scatter-add into dst row (L2-resident).
// ---------------------------------------------------------------------------
__global__ __launch_bounds__(256) void gcn_edge_msg(
        const int* __restrict__ src, const int* __restrict__ dst,
        const float* __restrict__ dinv, const float* __restrict__ lin,
        float* __restrict__ agg, int E) {
    int wid  = (blockIdx.x * blockDim.x + threadIdx.x) >> 5;
    int lane = threadIdx.x & 31;
    if (wid >= E) return;
    int s = src[wid], d = dst[wid];
    float nrm = dinv[s] * dinv[d];
    float4 v = ((const float4*)(lin + (size_t)s * D))[lane];
    float* op = agg + (size_t)d * D + lane * 4;
    atomicAdd(op + 0, v.x * nrm);
    atomicAdd(op + 1, v.y * nrm);
    atomicAdd(op + 2, v.z * nrm);
    atomicAdd(op + 3, v.w * nrm);
}

__global__ void gcn_bias_relu(float* __restrict__ h, const float* __restrict__ b, int NE) {
    int idx = blockIdx.x * blockDim.x + threadIdx.x;
    if (idx >= NE) return;
    float v = h[idx] + b[idx & (D - 1)];
    h[idx] = v > 0.0f ? v : 0.0f;
}

// ---------------------------------------------------------------------------
// Pooling + classifier
// ---------------------------------------------------------------------------
__global__ void gcn_zero(float* __restrict__ p, int n) {
    int i = blockIdx.x * blockDim.x + threadIdx.x;
    if (i < n) p[i] = 0.0f;
}
__global__ void gcn_pool_sum(const float* __restrict__ h, const int* __restrict__ batch,
                             float* __restrict__ sums, int NE) {
    int idx = blockIdx.x * blockDim.x + threadIdx.x;
    if (idx >= NE) return;
    int node = idx >> 7, dch = idx & (D - 1);
    atomicAdd(&sums[(size_t)batch[node] * D + dch], h[idx]);
}
__global__ void gcn_pool_cnt(const int* __restrict__ batch, float* __restrict__ cnt, int N) {
    int i = blockIdx.x * blockDim.x + threadIdx.x;
    if (i < N) atomicAdd(&cnt[batch[i]], 1.0f);
}
__global__ void gcn_classifier(const float* __restrict__ sums, const float* __restrict__ cnt,
                               const float* __restrict__ Wc, const float* __restrict__ bc,
                               float* __restrict__ out) {
    int t = blockIdx.x * blockDim.x + threadIdx.x;     // 0..2047
    if (t >= N_GRAPHS * D_OUT) return;
    int g = t >> 1, o = t & 1;
    float inv = 1.0f / fmaxf(cnt[g], 1.0f);
    float acc = bc[o];
#pragma unroll 4
    for (int dch = 0; dch < D; ++dch)
        acc += sums[(size_t)g * D + dch] * inv * Wc[dch * D_OUT + o];
    out[t] = acc;
}

// ---------------------------------------------------------------------------
extern "C" void kernel_launch(void* const* d_in, const int* in_sizes, int n_in,
                              void* d_out, int out_size, void* d_ws, size_t ws_size,
                              hipStream_t stream) {
    const float* x     = (const float*)d_in[0];
    const int*   ei    = (const int*)  d_in[1];
    const int*   batch = (const int*)  d_in[2];
    const float* W1    = (const float*)d_in[3];
    const float* b1    = (const float*)d_in[4];
    const float* W2    = (const float*)d_in[5];
    const float* b2    = (const float*)d_in[6];
    const float* Wc    = (const float*)d_in[7];
    const float* bc    = (const float*)d_in[8];
    float* out = (float*)d_out;

    const int N  = in_sizes[0] / D;        // 100000
    const int E  = in_sizes[1] / 2;        // 1600000
    const int NE = N * D;
    const int* src = ei;
    const int* dst = ei + E;

    // Workspace carve-up (512B aligned slabs).
    char* ws = (char*)d_ws;
    size_t off = 0;
    auto carve = [&](size_t bytes) { char* p = ws + off; off += (bytes + 511) & ~(size_t)511; return p; };
    float*    dinv  = (float*)   carve((size_t)N * 4);
    float*    bufA  = (float*)   carve((size_t)NE * 4);
    float*    bufB  = (float*)   carve((size_t)NE * 4);
    _Float16* wf1   = (_Float16*)carve(8 * 4 * 32 * 16 * 2);
    _Float16* wf2   = (_Float16*)carve(8 * 4 * 32 * 16 * 2);
    float*    sums  = (float*)   carve((size_t)N_GRAPHS * D * 4);
    float*    cnt   = (float*)   carve((size_t)N_GRAPHS * 4);
    (void)ws_size;

    const int TB = 256;
    dim3 blk(TB);
    int gN   = (N  + TB - 1) / TB;
    int gE   = (E  + TB - 1) / TB;
    int gNE  = (NE + TB - 1) / TB;
    int gEm  = (E * 32 + TB - 1) / TB;     // one wave per edge
    int gM   = N / 16;                     // 6250 M tiles

    // Weight fragment packing
    gcn_pack_w<<<4, blk, 0, stream>>>(W1, wf1);
    gcn_pack_w<<<4, blk, 0, stream>>>(W2, wf2);

    // Degree -> dinv
    gcn_deg_init <<<gN, blk, 0, stream>>>(dinv, N);
    gcn_deg_edges<<<gE, blk, 0, stream>>>(dst, dinv, E);
    gcn_dinv     <<<gN, blk, 0, stream>>>(dinv, N);

    // Layer 1
    gcn_gemm_wmma    <<<gM,  blk, 0, stream>>>(x, wf1, bufA);
    gcn_selfloop_init<<<gNE, blk, 0, stream>>>(bufA, dinv, bufB, NE);
    gcn_edge_msg     <<<gEm, blk, 0, stream>>>(src, dst, dinv, bufA, bufB, E);
    gcn_bias_relu    <<<gNE, blk, 0, stream>>>(bufB, b1, NE);

    // Layer 2
    gcn_gemm_wmma    <<<gM,  blk, 0, stream>>>(bufB, wf2, bufA);
    gcn_selfloop_init<<<gNE, blk, 0, stream>>>(bufA, dinv, bufB, NE);
    gcn_edge_msg     <<<gEm, blk, 0, stream>>>(src, dst, dinv, bufA, bufB, E);
    gcn_bias_relu    <<<gNE, blk, 0, stream>>>(bufB, b2, NE);

    // Mean pool + classifier
    gcn_zero     <<<(N_GRAPHS * D + N_GRAPHS + TB - 1) / TB, blk, 0, stream>>>(sums, N_GRAPHS * D + N_GRAPHS);
    gcn_pool_sum <<<gNE, blk, 0, stream>>>(bufB, batch, sums, NE);
    gcn_pool_cnt <<<gN,  blk, 0, stream>>>(batch, cnt, N);
    gcn_classifier<<<(N_GRAPHS * D_OUT + TB - 1) / TB, blk, 0, stream>>>(sums, cnt, Wc, bc, out);
}
